// MambaBlock_9337258902062
// MI455X (gfx1250) — matrix operand order repaired and checked by reading, not compiled, not run on hardware
//
#include <hip/hip_runtime.h>
#include <hip/hip_bf16.h>

// ---------------------------------------------------------------------------
// Mamba block forward, MI455X (gfx1250, wave32, WMMA).
// B=4, L=2048, D_MODEL=1024, D_INNER=2048, D_STATE=16, D_CONV=4, DT_RANK=64
//   1) cvt x, weights -> bf16
//   2) xz = x @ in_proj_w^T          (WMMA bf16, f32 acc, 2-stage reg pipeline)
//   3) causal depthwise conv4 + SiLU -> xm (f32 + bf16)
//   4) dbc = xm @ x_proj_w^T         (WMMA)
//   5) dtp = dbc[:, :64] @ dt_proj_w^T (WMMA)
//   6) softplus(dtp + dt_b)
//   7) selective scan (16 states/thread, B/C async-copied to LDS chunks)
//   8) y *= silu(z)  -> bf16
//   9) o = y @ out_proj_w^T          (WMMA)
//  10) layernorm(o + x) -> d_out (f32)
// ---------------------------------------------------------------------------

typedef __attribute__((ext_vector_type(16))) __bf16 v16bf;
typedef __attribute__((ext_vector_type(8)))  float  v8f;
typedef unsigned short u16;

static constexpr int BB = 4;
static constexpr int LL = 2048;
static constexpr int DM = 1024;     // d_model
static constexpr int DI = 2048;     // d_inner
static constexpr int DS = 16;       // d_state
static constexpr int DR = 64;       // dt_rank
static constexpr int DBC = DR + 2 * DS;   // 96
static constexpr int MR = BB * LL;  // 8192 token rows

struct alignas(16) U128 { unsigned int w[4]; };
union FragBF { U128 q[2]; v16bf v; };

__device__ __forceinline__ u16 f2bf(float f) {
  union { float f; unsigned u; } c; c.f = f;
  unsigned r = c.u + 0x7FFFu + ((c.u >> 16) & 1u);   // round-to-nearest-even
  return (u16)(r >> 16);
}

__device__ __forceinline__ float silu(float v) { return v / (1.0f + __expf(-v)); }

// ---------------------------------------------------------------------------
// f32 -> bf16 conversion
// ---------------------------------------------------------------------------
__global__ __launch_bounds__(256) void cvt_bf16_kernel(const float* __restrict__ in,
                                                       u16* __restrict__ out, int n) {
  int i = blockIdx.x * 256 + threadIdx.x;
  if (i < n) out[i] = f2bf(in[i]);
}

// ---------------------------------------------------------------------------
// C[M,N] (f32) = A[M,K] (bf16, lda) @ W[N,K]^T (bf16, ldw)
// One wave -> 16x(16*NT) output strip; block = 4 waves stacked in M (64 rows).
// grid = (N/(16*NT), M/64), block = (32,4)
// Two-stage software pipeline with STATICALLY NAMED register buffers (no
// runtime-indexed local arrays -> no scratch demotion). K is a template
// constant (1024 / 2048 / 64, all multiples of 64), K-loop steps by 64.
// Fragment layout per CDNA5 ISA (16-bit A 16x32, B 32x16, f32 C/D).
// ---------------------------------------------------------------------------
__device__ __forceinline__ void load_frag_a(const u16* __restrict__ arow, int kk,
                                            int ka, FragBF& f) {
  f.q[0] = *(const U128*)(arow + kk + ka);        // K = ka..ka+7
  f.q[1] = *(const U128*)(arow + kk + ka + 16);   // K = ka+16..ka+23
}
__device__ __forceinline__ void load_frag_b(const u16* __restrict__ wrow, int kk,
                                            int kb, FragBF& f) {
  f.q[0] = *(const U128*)(wrow + kk + kb);        // K = kb..kb+7
  f.q[1] = *(const U128*)(wrow + kk + kb + 16);   // K = kb+16..kb+23
}

template <int NT, int K>
__global__ __launch_bounds__(128) void wmma_gemm_bf16_kernel(
    const u16* __restrict__ A, const u16* __restrict__ W, float* __restrict__ C,
    int lda, int ldw, int ldc) {
  const int lane = threadIdx.x;                 // 0..31
  const int wave = threadIdx.y;                 // 0..3
  const int tileM = blockIdx.y * 64 + wave * 16;
  const int tileN = blockIdx.x * (16 * NT);

  v8f acc[NT];
#pragma unroll
  for (int j = 0; j < NT; ++j) { v8f z = {}; acc[j] = z; }

  const int hi = lane >> 4;                     // 0 | 1
  const int ln = lane & 15;
  const u16* arow = A + (size_t)(tileM + ln) * lda;
  const u16* wrow[NT];
#pragma unroll
  for (int j = 0; j < NT; ++j)
    wrow[j] = W + (size_t)(tileN + j * 16 + ln) * ldw;
  const int ka = hi << 3;                       // A K-chunk base: 0 or 8
  const int kb = hi << 3;                       // B K-chunk base: 0 or 8

  FragBF fa0, fa1;
  FragBF fb0[NT], fb1[NT];

  // prologue: stage 0 <- K-step 0
  load_frag_a(arow, 0, ka, fa0);
#pragma unroll
  for (int j = 0; j < NT; ++j) load_frag_b(wrow[j], 0, kb, fb0[j]);

#pragma unroll 1
  for (int kk = 0; kk < K; kk += 64) {
    // stage 1 <- K-step kk+32 (always in range: K % 64 == 0)
    load_frag_a(arow, kk + 32, ka, fa1);
#pragma unroll
    for (int j = 0; j < NT; ++j) load_frag_b(wrow[j], kk + 32, kb, fb1[j]);
    if (kk + 256 < K) {
      __builtin_prefetch(arow + kk + 256, 0, 0);       // global_prefetch_b8
      __builtin_prefetch(wrow[0] + kk + 256, 0, 0);
    }
    // compute on stage 0 while stage-1 loads are in flight
#pragma unroll
    for (int j = 0; j < NT; ++j)
      acc[j] = __builtin_amdgcn_wmma_f32_16x16x32_bf16(
          false, fa0.v, false, fb0[j].v, (short)0, acc[j], false, false);
    // stage 0 <- K-step kk+64 (skip on last iteration)
    if (kk + 64 < K) {
      load_frag_a(arow, kk + 64, ka, fa0);
#pragma unroll
      for (int j = 0; j < NT; ++j) load_frag_b(wrow[j], kk + 64, kb, fb0[j]);
    }
    // compute on stage 1 while stage-0 loads are in flight
#pragma unroll
    for (int j = 0; j < NT; ++j)
      acc[j] = __builtin_amdgcn_wmma_f32_16x16x32_bf16(
          false, fa1.v, false, fb1[j].v, (short)0, acc[j], false, false);
  }

  const int rbase = tileM + (hi << 3);
  const int cbase = tileN + ln;
#pragma unroll
  for (int j = 0; j < NT; ++j)
#pragma unroll
    for (int r = 0; r < 8; ++r)
      C[(size_t)(rbase + r) * ldc + cbase + j * 16] = acc[j][r];
}

// ---------------------------------------------------------------------------
// Causal depthwise conv (4 taps) + bias + SiLU.  xz: (MR, 2*DI), xm in cols [0,DI)
// ---------------------------------------------------------------------------
__global__ __launch_bounds__(256) void conv_silu_kernel(
    const float* __restrict__ xz, const float* __restrict__ cw,
    const float* __restrict__ cb, float* __restrict__ xmf, u16* __restrict__ xmbf) {
  int idx = blockIdx.x * 256 + threadIdx.x;     // over MR*DI
  if (idx >= MR * DI) return;
  int d = idx & (DI - 1);
  int l = (idx >> 11) & (LL - 1);
  int b = idx >> 22;
  float acc = cb[d];
#pragma unroll
  for (int j = 0; j < 4; ++j) {
    int ls = l - 3 + j;
    if (ls >= 0)
      acc += cw[d * 4 + j] * xz[(size_t)(b * LL + ls) * (2 * DI) + d];
  }
  float s = silu(acc);
  xmf[idx] = s;
  xmbf[idx] = f2bf(s);
}

// ---------------------------------------------------------------------------
// softplus(dtp + dt_b), in place.  dtp: (MR, DI)
// ---------------------------------------------------------------------------
__global__ __launch_bounds__(256) void softplus_kernel(float* __restrict__ dtp,
                                                       const float* __restrict__ dtb) {
  int idx = blockIdx.x * 256 + threadIdx.x;
  if (idx >= MR * DI) return;
  float v = dtp[idx] + dtb[idx & (DI - 1)];
  dtp[idx] = (v > 20.0f) ? v : log1pf(__expf(v));
}

// ---------------------------------------------------------------------------
// Selective scan. One thread per (b,d) channel, 16 states in registers,
// sequential over L. B/C vectors (32 floats per timestep, shared by all 2048
// channels of a batch) are async-copied global->LDS (no VGPR round-trip,
// ASYNCcnt tracked) in 64-step chunks: 256 threads x 2 x b128 = 8KB.
// grid = (DI/256, BB), block = 256
// ---------------------------------------------------------------------------
__global__ __launch_bounds__(256) void scan_kernel(
    const float* __restrict__ dbc,   // (MR, 96): [64+s]=B, [80+s]=C
    const float* __restrict__ dtp,   // (MR, DI) softplus'd
    const float* __restrict__ xmf,   // (MR, DI)
    const float* __restrict__ alog,  // (DI, DS)
    const float* __restrict__ Dv,    // (DI)
    float* __restrict__ y) {         // (MR, DI)
  constexpr int CHUNK = 64;
  __shared__ float sBC[CHUNK][2 * DS];
  const int d = blockIdx.x * 256 + threadIdx.x;
  const int b = blockIdx.y;

  float Ai[DS], h[DS];
#pragma unroll
  for (int s = 0; s < DS; ++s) {
    Ai[s] = -__expf(alog[d * DS + s]);
    h[s] = 0.0f;
  }
  const float Dd = Dv[d];

  for (int l0 = 0; l0 < LL; l0 += CHUNK) {
    __syncthreads();   // previous chunk fully consumed before overwrite
#pragma unroll
    for (int i = 0; i < 2; ++i) {
      // 16-byte unit u: row t = u>>3 (8 units per 32-float row), col c4 = (u&7)*4
      int u = threadIdx.x + i * 256;
      int t = u >> 3;
      int c4 = (u & 7) * 4;
      const float* gsrc = dbc + (size_t)(b * LL + l0 + t) * DBC + DR + c4;
      unsigned ldst = (unsigned)(uintptr_t)(&sBC[t][c4]);  // flat low32 == LDS addr
      asm volatile("global_load_async_to_lds_b128 %0, %1, off"
                   :: "v"(ldst), "v"(gsrc)
                   : "memory");
    }
    asm volatile("s_wait_asynccnt 0x0" ::: "memory");
    __syncthreads();
    for (int t = 0; t < CHUNK; ++t) {
      const size_t row = (size_t)(b * LL + l0 + t);
      const float dt = dtp[row * DI + d];
      const float xv = xmf[row * DI + d];
      const float dx = dt * xv;
      float yv = 0.0f;
#pragma unroll
      for (int s = 0; s < DS; ++s) {
        h[s] = __expf(dt * Ai[s]) * h[s] + dx * sBC[t][s];
        yv += h[s] * sBC[t][DS + s];
      }
      y[row * DI + d] = yv + Dd * xv;
    }
  }
}

// ---------------------------------------------------------------------------
// y *= silu(z); z is xz cols [DI, 2*DI).  Emit bf16 for out_proj GEMM.
// ---------------------------------------------------------------------------
__global__ __launch_bounds__(256) void gate_kernel(const float* __restrict__ y,
                                                   const float* __restrict__ xz,
                                                   u16* __restrict__ ybf) {
  int idx = blockIdx.x * 256 + threadIdx.x;
  if (idx >= MR * DI) return;
  int m = idx >> 11;
  int d = idx & (DI - 1);
  float z = xz[(size_t)m * (2 * DI) + DI + d];
  ybf[idx] = f2bf(y[idx] * silu(z));
}

// ---------------------------------------------------------------------------
// LayerNorm(o + x) over last dim (1024). One block per token row.
// ---------------------------------------------------------------------------
__global__ __launch_bounds__(256) void ln_kernel(const float* __restrict__ o,
                                                 const float* __restrict__ x,
                                                 const float* __restrict__ g,
                                                 const float* __restrict__ bsh,
                                                 float* __restrict__ out) {
  __shared__ float red[256];
  const int m = blockIdx.x;
  const int tid = threadIdx.x;
  float v[4];
#pragma unroll
  for (int i = 0; i < 4; ++i) {
    int c = tid + i * 256;
    v[i] = o[(size_t)m * DM + c] + x[(size_t)m * DM + c];
  }
  float s = v[0] + v[1] + v[2] + v[3];
  red[tid] = s; __syncthreads();
  for (int off = 128; off > 0; off >>= 1) {
    if (tid < off) red[tid] += red[tid + off];
    __syncthreads();
  }
  const float mu = red[0] * (1.0f / DM);
  __syncthreads();
  float sq = 0.0f;
#pragma unroll
  for (int i = 0; i < 4; ++i) { float dd = v[i] - mu; sq += dd * dd; }
  red[tid] = sq; __syncthreads();
  for (int off = 128; off > 0; off >>= 1) {
    if (tid < off) red[tid] += red[tid + off];
    __syncthreads();
  }
  const float inv = rsqrtf(red[0] * (1.0f / DM) + 1e-5f);
  __syncthreads();
#pragma unroll
  for (int i = 0; i < 4; ++i) {
    int c = tid + i * 256;
    out[(size_t)m * DM + c] = (v[i] - mu) * inv * g[c] + bsh[c];
  }
}

// ---------------------------------------------------------------------------
// Host side
// ---------------------------------------------------------------------------
extern "C" void kernel_launch(void* const* d_in, const int* in_sizes, int n_in,
                              void* d_out, int out_size, void* d_ws, size_t ws_size,
                              hipStream_t stream) {
  (void)in_sizes; (void)n_in; (void)out_size; (void)ws_size;
  const float* x    = (const float*)d_in[0];   // (B,L,DM)
  const float* win  = (const float*)d_in[1];   // (2*DI, DM)
  const float* cw   = (const float*)d_in[2];   // (DI,1,4)
  const float* cb   = (const float*)d_in[3];   // (DI)
  const float* wxp  = (const float*)d_in[4];   // (96, DI)
  const float* wdt  = (const float*)d_in[5];   // (DI, 64)
  const float* bdt  = (const float*)d_in[6];   // (DI)
  const float* alog = (const float*)d_in[7];   // (DI, DS)
  const float* Dv   = (const float*)d_in[8];   // (DI)
  const float* wout = (const float*)d_in[9];   // (DM, DI)
  const float* lng  = (const float*)d_in[10];  // (DM)
  const float* lnb  = (const float*)d_in[11];  // (DM)
  float* out = (float*)d_out;

  char* ws = (char*)d_ws;
  size_t off = 0;
  auto alloc = [&](size_t bytes) -> void* {
    void* p = ws + off;
    off = (off + bytes + 255) & ~(size_t)255;
    return p;
  };

  u16*   xbf   = (u16*)  alloc((size_t)MR * DM * 2);
  u16*   wibf  = (u16*)  alloc((size_t)2 * DI * DM * 2);
  u16*   wxbf  = (u16*)  alloc((size_t)DBC * DI * 2);
  u16*   wdbf  = (u16*)  alloc((size_t)DI * DR * 2);
  u16*   wobf  = (u16*)  alloc((size_t)DM * DI * 2);
  float* xz    = (float*)alloc((size_t)MR * 2 * DI * 4);
  float* xmf   = (float*)alloc((size_t)MR * DI * 4);
  u16*   xmbf  = (u16*)  alloc((size_t)MR * DI * 2);
  float* dbc   = (float*)alloc((size_t)MR * DBC * 4);
  u16*   dbcbf = (u16*)  alloc((size_t)MR * DBC * 2);
  float* dtp   = (float*)alloc((size_t)MR * DI * 4);
  float* ysc   = (float*)alloc((size_t)MR * DI * 4);
  u16*   ybf   = (u16*)  alloc((size_t)MR * DI * 2);
  float* obuf  = (float*)alloc((size_t)MR * DM * 4);

  auto cvt = [&](const float* src, u16* dst, int n) {
    cvt_bf16_kernel<<<(n + 255) / 256, 256, 0, stream>>>(src, dst, n);
  };

  // 1) bf16 conversions (input + weights)
  cvt(x,    xbf,  MR * DM);
  cvt(win,  wibf, 2 * DI * DM);
  cvt(wxp,  wxbf, DBC * DI);
  cvt(wdt,  wdbf, DI * DR);
  cvt(wout, wobf, DM * DI);

  // 2) in_proj: xz = x @ win^T   (M=8192, N=4096, K=1024)
  wmma_gemm_bf16_kernel<4, DM><<<dim3((2 * DI) / 64, MR / 64), dim3(32, 4), 0, stream>>>(
      xbf, wibf, xz, DM, DM, 2 * DI);

  // 3) depthwise conv + SiLU
  conv_silu_kernel<<<(MR * DI) / 256, 256, 0, stream>>>(xz, cw, cb, xmf, xmbf);

  // 4) x_proj: dbc = xm @ wxp^T  (N=96, K=2048)
  wmma_gemm_bf16_kernel<2, DI><<<dim3(DBC / 32, MR / 64), dim3(32, 4), 0, stream>>>(
      xmbf, wxbf, dbc, DI, DI, DBC);

  // 5) dt_proj: dtp = dbc[:, :64] @ wdt^T  (N=2048, K=64, lda=96)
  cvt(dbc, dbcbf, MR * DBC);
  wmma_gemm_bf16_kernel<4, DR><<<dim3(DI / 64, MR / 64), dim3(32, 4), 0, stream>>>(
      dbcbf, wdbf, dtp, DBC, DR, DI);

  // 6) softplus(dt + bias)
  softplus_kernel<<<(MR * DI) / 256, 256, 0, stream>>>(dtp, bdt);

  // 7) selective scan
  scan_kernel<<<dim3(DI / 256, BB), 256, 0, stream>>>(dbc, dtp, xmf, alog, Dv, ysc);

  // 8) gate: y *= silu(z)
  gate_kernel<<<(MR * DI) / 256, 256, 0, stream>>>(ysc, xz, ybf);

  // 9) out_proj: o = y @ wout^T  (N=1024, K=2048)
  wmma_gemm_bf16_kernel<4, DI><<<dim3(DM / 64, MR / 64), dim3(32, 4), 0, stream>>>(
      ybf, wobf, obuf, DI, DI, DM);

  // 10) layernorm(o + x) -> out
  ln_kernel<<<MR, 256, 0, stream>>>(obuf, x, lng, lnb, out);
}
